// Tagger_44074954391734
// MI455X (gfx1250) — matrix operand order repaired and checked by reading, not compiled, not run on hardware
//
#include <hip/hip_runtime.h>
#include <hip/hip_bf16.h>
#include <cstdint>
#include <cstddef>

// ---------------------------------------------------------------------------
// Problem constants (from reference): V=50000, I=512, H=1024, OUT=2, P=256,
// L=64, NH=32. seq_len==1 attention => softmax==1 => ctx == v projection.
// ---------------------------------------------------------------------------
#define P_ 256
#define L_ 64
#define I_ 512
#define H_ 1024

typedef __attribute__((ext_vector_type(16))) __bf16 v16bf;
typedef __attribute__((ext_vector_type(8)))  float  v8f;
typedef __attribute__((ext_vector_type(4)))  int    v4i;

__device__ __forceinline__ float sigf(float x) { return 1.0f / (1.0f + __expf(-x)); }
// fragment F -> element offset of this lane's 16-bf16 chunk
__device__ __forceinline__ size_t frag_off(size_t F, int lane) {
    return (F * 32 + (size_t)lane) * 16;
}

// ---- CDNA5 async global->LDS staging (guarded; safe fallback) -------------
#if defined(__has_builtin)
#if __has_builtin(__builtin_amdgcn_global_load_async_to_lds_b128)
#define HAVE_ASYNC_LDS 1
#else
#define HAVE_ASYNC_LDS 0
#endif
#else
#define HAVE_ASYNC_LDS 0
#endif

#if HAVE_ASYNC_LDS
// signature (from clang diagnostic): (v4i addrspace(1)*, v4i addrspace(3)*, imm offset, imm cpol)
#define STAGE_COPY(s_, d_) __builtin_amdgcn_global_load_async_to_lds_b128( \
    (__attribute__((address_space(1))) v4i*)(s_),                          \
    (__attribute__((address_space(3))) v4i*)(d_), 0, 0)
#define ASYNC_WAIT_LE1() asm volatile("s_wait_asynccnt 0x1" ::: "memory")
#define ASYNC_WAIT_0()   asm volatile("s_wait_asynccnt 0x0" ::: "memory")
#else
#define STAGE_COPY(s_, d_) (*(uint4*)(d_) = *(const uint4*)(s_))
#define ASYNC_WAIT_LE1() ((void)0)
#define ASYNC_WAIT_0()   ((void)0)
#endif

#define WM4(A_, B0_, B1_, B2_, B3_) do { \
    acc0 = __builtin_amdgcn_wmma_f32_16x16x32_bf16(false, A_, false, B0_, (short)0, acc0, false, false); \
    acc1 = __builtin_amdgcn_wmma_f32_16x16x32_bf16(false, A_, false, B1_, (short)0, acc1, false, false); \
    acc2 = __builtin_amdgcn_wmma_f32_16x16x32_bf16(false, A_, false, B2_, (short)0, acc2, false, false); \
    acc3 = __builtin_amdgcn_wmma_f32_16x16x32_bf16(false, A_, false, B3_, (short)0, acc3, false, false); } while (0)

// ---------------------------------------------------------------------------
// LDS-staged WMMA strip. All 8 waves of the block consume the SAME 4 B
// fragments per k-tile, so the block stages them in LDS once (double
// buffered, async when available). A fragments are per-wave, kept in a
// 2-deep register rotation. KT is even (16 or 32).
// Staging role: thread tid stages 16B chunk (tid&63) of gate (tid>>6).
// ---------------------------------------------------------------------------
template<int KT>
__device__ __forceinline__ void strip_lds(
    const __bf16* __restrict__ ap,   // this wave's A frag base (lane folded in)
    const __bf16* __restrict__ Bf,   // B fragment array
    size_t sfb,                      // staging frag base for this thread's gate
    int lane, int sg, int sc,
    __bf16 (&sB)[2][4][512],
    v8f& acc0, v8f& acc1, v8f& acc2, v8f& acc3)
{
#define STAGE(kt_, buf_) do { \
    const __bf16* src_ = Bf + (sfb + (size_t)(kt_)) * 512 + (size_t)sc * 8; \
    __bf16* dst_ = &sB[buf_][sg][sc * 8]; \
    STAGE_COPY(src_, dst_); } while (0)

    v16bf aA = *(const v16bf*)(ap);
    v16bf aB = *(const v16bf*)(ap + 512);
    STAGE(0, 0);
    STAGE(1, 1);
#pragma unroll
    for (int kt = 0; kt < KT; kt += 2) {
        {   // even phase: buffer 0
            ASYNC_WAIT_LE1();
            __syncthreads();
            v16bf b0 = *(const v16bf*)&sB[0][0][lane * 16];
            v16bf b1 = *(const v16bf*)&sB[0][1][lane * 16];
            v16bf b2 = *(const v16bf*)&sB[0][2][lane * 16];
            v16bf b3 = *(const v16bf*)&sB[0][3][lane * 16];
            WM4(aA, b0, b1, b2, b3);
            __syncthreads();
            if (kt + 2 < KT) {
                STAGE(kt + 2, 0);
                aA = *(const v16bf*)(ap + (size_t)(kt + 2) * 512);
            }
        }
        {   // odd phase: buffer 1
            if (kt + 1 == KT - 1) { ASYNC_WAIT_0(); } else { ASYNC_WAIT_LE1(); }
            __syncthreads();
            v16bf b0 = *(const v16bf*)&sB[1][0][lane * 16];
            v16bf b1 = *(const v16bf*)&sB[1][1][lane * 16];
            v16bf b2 = *(const v16bf*)&sB[1][2][lane * 16];
            v16bf b3 = *(const v16bf*)&sB[1][3][lane * 16];
            WM4(aB, b0, b1, b2, b3);
            __syncthreads();
            if (kt + 3 < KT) {
                STAGE(kt + 3, 1);
                aB = *(const v16bf*)(ap + (size_t)(kt + 3) * 512);
            }
        }
    }
#undef STAGE
}

// ---------------------------------------------------------------------------
// Register-pipelined strip (global loads only) for the small post GEMMs.
// ---------------------------------------------------------------------------
template<int KT>
__device__ __forceinline__ void wmma_strip(
    const __bf16* __restrict__ ap,
    const __bf16* __restrict__ p0, const __bf16* __restrict__ p1,
    const __bf16* __restrict__ p2, const __bf16* __restrict__ p3,
    v8f& acc0, v8f& acc1, v8f& acc2, v8f& acc3)
{
#define LDG(A_, B0_, B1_, B2_, B3_, KTi) do { size_t o_ = (size_t)(KTi) * 512; \
    A_  = *(const v16bf*)(ap + o_); \
    B0_ = *(const v16bf*)(p0 + o_); \
    B1_ = *(const v16bf*)(p1 + o_); \
    B2_ = *(const v16bf*)(p2 + o_); \
    B3_ = *(const v16bf*)(p3 + o_); } while (0)

    v16bf aA, c0A, c1A, c2A, c3A;
    v16bf aB, c0B, c1B, c2B, c3B;
    LDG(aA, c0A, c1A, c2A, c3A, 0);
    LDG(aB, c0B, c1B, c2B, c3B, 1);
#pragma unroll
    for (int kt = 0; kt < KT - 2; kt += 2) {
        WM4(aA, c0A, c1A, c2A, c3A);
        LDG(aA, c0A, c1A, c2A, c3A, kt + 2);
        WM4(aB, c0B, c1B, c2B, c3B);
        LDG(aB, c0B, c1B, c2B, c3B, kt + 3);
    }
    WM4(aA, c0A, c1A, c2A, c3A);
    WM4(aB, c0B, c1B, c2B, c3B);
#undef LDG
}

// ---------------------------------------------------------------------------
// Pack kernels: build WMMA fragment-ordered bf16 operands.
// A-frag (16x32, MxK): lane<16 -> row=lane,     k = {0..7, 16..23} of k-tile
//                      lane>=16-> row=lane-16,  k = {8..15, 24..31}
// B-frag (32x16, KxN): lane<16 -> col=lane,     k = {0..7, 16..23}
//                      lane>=16-> col=lane-16,  k = {8..15, 24..31}
// ---------------------------------------------------------------------------

// Embedding gather directly into A-frag layout: Xf[t][mTile][kTile][lane][16]
__global__ void embed_pack_kernel(const int* __restrict__ paths,
                                  const float* __restrict__ emb,
                                  __bf16* __restrict__ Xf)
{
    int tid  = blockIdx.x * blockDim.x + threadIdx.x;  // L*16*16*32 = 524288
    int lane = tid & 31;
    int kt   = (tid >> 5) & 15;   // 16 k-tiles (K = I = 512)
    int mt   = (tid >> 9) & 15;   // 16 m-tiles (M = P = 256)
    int t    = tid >> 13;         // 0..63
    int row  = mt * 16 + (lane & 15);
    int k0   = kt * 32 + (lane >> 4) * 8;
    int tok  = paths[row * L_ + t];
    const float* src = emb + (size_t)tok * I_ + k0;
    v16bf v;
#pragma unroll
    for (int i = 0; i < 8; ++i) {
        v[i]     = (__bf16)src[i];
        v[8 + i] = (__bf16)src[16 + i];
    }
    *(v16bf*)(Xf + frag_off((size_t)t * 256 + mt * 16 + kt, lane)) = v;
}

// f32 row-major [MT*16 x KT*32] -> A-frag layout [mt][kt][lane][16]
__global__ void pack_a_kernel(const float* __restrict__ src,
                              __bf16* __restrict__ Af, int KT)
{
    int tid  = blockIdx.x * blockDim.x + threadIdx.x;  // MT*KT*32
    int lane = tid & 31;
    int f    = tid >> 5;
    int kt   = f % KT;
    int mt   = f / KT;
    int row  = mt * 16 + (lane & 15);
    int k0   = kt * 32 + (lane >> 4) * 8;
    const float* s = src + (size_t)row * ((size_t)KT * 32) + k0;
    v16bf v;
#pragma unroll
    for (int i = 0; i < 8; ++i) {
        v[i]     = (__bf16)s[i];
        v[8 + i] = (__bf16)s[16 + i];
    }
    *(v16bf*)(Af + frag_off((size_t)f, lane)) = v;
}

// bf16 row-major -> A-frag layout (pure shuffle, used for the h state)
__global__ void pack_a_bf16_kernel(const __bf16* __restrict__ src,
                                   __bf16* __restrict__ Af, int KT)
{
    int tid  = blockIdx.x * blockDim.x + threadIdx.x;  // MT*KT*32
    int lane = tid & 31;
    int f    = tid >> 5;
    int kt   = f % KT;
    int mt   = f / KT;
    int row  = mt * 16 + (lane & 15);
    int k0   = kt * 32 + (lane >> 4) * 8;
    const __bf16* s = src + (size_t)row * ((size_t)KT * 32) + k0;
    v16bf v;
#pragma unroll
    for (int i = 0; i < 8; ++i) {
        v[i]     = s[i];
        v[8 + i] = s[16 + i];
    }
    *(v16bf*)(Af + frag_off((size_t)f, lane)) = v;
}

// Weight W [N x K] row-major (torch layout; B[k][n] = W[n][k])
// -> B-frag layout [nt][kt][lane][16]
__global__ void pack_b_kernel(const float* __restrict__ W,
                              __bf16* __restrict__ Bf, int KT)
{
    int tid  = blockIdx.x * blockDim.x + threadIdx.x;  // NT*KT*32
    int lane = tid & 31;
    int f    = tid >> 5;
    int kt   = f % KT;
    int nt   = f / KT;
    int n    = nt * 16 + (lane & 15);
    int k0   = kt * 32 + (lane >> 4) * 8;
    const float* s = W + (size_t)n * ((size_t)KT * 32) + k0;
    v16bf v;
#pragma unroll
    for (int i = 0; i < 16; ++i) v[i] = (__bf16)s[i < 8 ? i : i + 8];
    *(v16bf*)(Bf + frag_off((size_t)f, lane)) = v;
}

// ---------------------------------------------------------------------------
// Fused LSTM step: gates GEMM (bf16 WMMA, f32 accum, LDS-staged B) + bias +
// cell update. Grid: 128 blocks x 256 thr; blockIdx = mGroup(2) x jGroup(64).
// Wave w: mTile = mGroup*8+w; accumulators are gates i,f,g,o for hidden
// columns jGroup*16..+15 (N-tiles jGroup, 64+jGroup, 128+jGroup, 192+jGroup).
// ---------------------------------------------------------------------------
__global__ void __launch_bounds__(256) lstm_step_kernel(
    const __bf16* __restrict__ Xf,   const __bf16* __restrict__ Hf,
    const __bf16* __restrict__ WihF, const __bf16* __restrict__ WhhF,
    const float* __restrict__ b_ih,  const float* __restrict__ b_hh,
    const int* __restrict__ lengths,
    float* __restrict__ c_state, __bf16* __restrict__ h_row,
    float* __restrict__ final_h, int t)
{
    __shared__ __attribute__((aligned(16))) __bf16 sB[2][4][512];  // 8KB double buffer

    const int tid    = threadIdx.x;
    const int lane   = tid & 31;
    const int wave   = tid >> 5;
    const int mGroup = blockIdx.x & 1;
    const int jGroup = blockIdx.x >> 1;     // 0..63
    const int mTile  = mGroup * 8 + wave;   // 0..15
    const int jBase  = jGroup * 16;
    const int sg     = tid >> 6;            // staging gate 0..3
    const int sc     = tid & 63;            // staging 16B chunk

    v8f acc0 = {0,0,0,0,0,0,0,0};
    v8f acc1 = {0,0,0,0,0,0,0,0};
    v8f acc2 = {0,0,0,0,0,0,0,0};
    v8f acc3 = {0,0,0,0,0,0,0,0};

    // ---- x_t @ W_ih^T : K = 512 (16 k-tiles) ----
    strip_lds<16>(
        Xf + frag_off((size_t)t * 256 + (size_t)mTile * 16, lane),
        WihF, (size_t)(sg * 64 + jGroup) * 16,
        lane, sg, sc, sB, acc0, acc1, acc2, acc3);

    // ---- h_{t-1} @ W_hh^T : K = 1024 (32 k-tiles) ----
    strip_lds<32>(
        Hf + frag_off((size_t)mTile * 32, lane),
        WhhF, (size_t)(sg * 64 + jGroup) * 32,
        lane, sg, sc, sB, acc0, acc1, acc2, acc3);

    // ---- fused LSTM cell epilogue (torch gate order i,f,g,o) ----
    // C layout: vgpr r, lane<16 -> (m=r, n=lane); lane>=16 -> (m=8+r, n=lane-16)
    const int hi  = lane >> 4;
    const int col = lane & 15;
    const int j   = jBase + col;
    const float bi = b_ih[j]          + b_hh[j];
    const float bf = b_ih[H_ + j]     + b_hh[H_ + j];
    const float bg = b_ih[2 * H_ + j] + b_hh[2 * H_ + j];
    const float bo = b_ih[3 * H_ + j] + b_hh[3 * H_ + j];
#pragma unroll
    for (int r = 0; r < 8; ++r) {
        const int p = mTile * 16 + hi * 8 + r;
        const size_t idx = (size_t)p * H_ + j;
        float gi = acc0[r] + bi;
        float gf = acc1[r] + bf;
        float gg = acc2[r] + bg;
        float go = acc3[r] + bo;
        float cn = sigf(gf) * c_state[idx] + sigf(gi) * tanhf(gg);
        float h  = sigf(go) * tanhf(cn);
        c_state[idx] = cn;
        h_row[idx]   = (__bf16)h;           // bf16 state for next step's GEMM
        if (lengths[p] == t + 1) final_h[idx] = h;  // h at the path's true length
    }
}

// ---------------------------------------------------------------------------
// Generic M=256, K=1024 GEMM + bias: out[256 x N] = Afrags * Bfrags + bias
// Grid: (2 mGroups) x (N/64 nGroups), 256 threads (8 waves).
// ---------------------------------------------------------------------------
__global__ void __launch_bounds__(256) gemm_bias_kernel(
    const __bf16* __restrict__ Af, const __bf16* __restrict__ Bf,
    const float* __restrict__ bias, float* __restrict__ out, int N)
{
    const int lane   = threadIdx.x & 31;
    const int wave   = threadIdx.x >> 5;
    const int mGroup = blockIdx.x & 1;
    const int nGroup = blockIdx.x >> 1;
    const int mTile  = mGroup * 8 + wave;
    const int KT     = 32;   // K = 1024 everywhere on this path

    v8f acc0 = {0,0,0,0,0,0,0,0};
    v8f acc1 = {0,0,0,0,0,0,0,0};
    v8f acc2 = {0,0,0,0,0,0,0,0};
    v8f acc3 = {0,0,0,0,0,0,0,0};

    wmma_strip<32>(
        Af + frag_off((size_t)mTile * KT, lane),
        Bf + frag_off((size_t)(nGroup * 4 + 0) * KT, lane),
        Bf + frag_off((size_t)(nGroup * 4 + 1) * KT, lane),
        Bf + frag_off((size_t)(nGroup * 4 + 2) * KT, lane),
        Bf + frag_off((size_t)(nGroup * 4 + 3) * KT, lane),
        acc0, acc1, acc2, acc3);

    const int hi  = lane >> 4;
    const int col = lane & 15;
    const int n0 = (nGroup * 4 + 0) * 16 + col;
    const int n1 = (nGroup * 4 + 1) * 16 + col;
    const int n2 = (nGroup * 4 + 2) * 16 + col;
    const int n3 = (nGroup * 4 + 3) * 16 + col;
    const float bv0 = bias[n0], bv1 = bias[n1], bv2 = bias[n2], bv3 = bias[n3];
#pragma unroll
    for (int r = 0; r < 8; ++r) {
        const size_t m = (size_t)(mTile * 16 + hi * 8 + r);
        out[m * N + n0] = acc0[r] + bv0;
        out[m * N + n1] = acc1[r] + bv1;
        out[m * N + n2] = acc2[r] + bv2;
        out[m * N + n3] = acc3[r] + bv3;
    }
}

// max over the P=256 paths (MaxPool2d((P,1)))
__global__ void maxpool_kernel(const float* __restrict__ a, float* __restrict__ pooled)
{
    int j = blockIdx.x * blockDim.x + threadIdx.x;
    if (j >= H_) return;
    float m = a[j];
    for (int p = 1; p < P_; ++p) m = fmaxf(m, a[(size_t)p * H_ + j]);
    pooled[j] = m;
}

// out = sigmoid(pooled @ W_lin^T + b_lin), OUT = 2
__global__ void final_out_kernel(const float* __restrict__ pooled,
                                 const float* __restrict__ W_lin,
                                 const float* __restrict__ b_lin,
                                 float* __restrict__ out)
{
    __shared__ float s0[128];
    __shared__ float s1[128];
    int tid = threadIdx.x;
    float a0 = 0.f, a1 = 0.f;
    for (int k = tid; k < H_; k += 128) {
        float p = pooled[k];
        a0 += p * W_lin[k];
        a1 += p * W_lin[H_ + k];
    }
    s0[tid] = a0; s1[tid] = a1;
    __syncthreads();
    for (int s = 64; s > 0; s >>= 1) {
        if (tid < s) { s0[tid] += s0[tid + s]; s1[tid] += s1[tid + s]; }
        __syncthreads();
    }
    if (tid == 0) {
        out[0] = sigf(s0[0] + b_lin[0]);
        out[1] = sigf(s1[0] + b_lin[1]);
    }
}

// ---------------------------------------------------------------------------
extern "C" void kernel_launch(void* const* d_in, const int* in_sizes, int n_in,
                              void* d_out, int out_size, void* d_ws, size_t ws_size,
                              hipStream_t stream)
{
    (void)in_sizes; (void)n_in; (void)out_size; (void)ws_size;
    const int*   paths   = (const int*)  d_in[0];
    const int*   lengths = (const int*)  d_in[1];
    const float* emb     = (const float*)d_in[2];
    const float* W_ih    = (const float*)d_in[3];
    const float* W_hh    = (const float*)d_in[4];
    const float* b_ih    = (const float*)d_in[5];
    const float* b_hh    = (const float*)d_in[6];
    const float* W_in    = (const float*)d_in[7];
    const float* b_in    = (const float*)d_in[8];
    const float* W_out   = (const float*)d_in[9];
    const float* b_out   = (const float*)d_in[10];
    const float* W_lin   = (const float*)d_in[11];
    const float* b_lin   = (const float*)d_in[12];
    float* out = (float*)d_out;

    uint8_t* ws = (uint8_t*)d_ws;
    size_t off = 0;
    auto carve = [&](size_t bytes) -> void* {
        off = (off + 255) & ~(size_t)255;
        void* p = ws + off;
        off += bytes;
        return p;
    };

    __bf16* WihF   = (__bf16*)carve((size_t)4 * H_ * I_ * 2);   // 4MB
    __bf16* WhhF   = (__bf16*)carve((size_t)4 * H_ * H_ * 2);   // 8MB
    __bf16* WvF    = (__bf16*)carve((size_t)H_ * H_ * 2);       // 2MB
    __bf16* WoutF  = (__bf16*)carve((size_t)H_ * H_ * 2);       // 2MB
    __bf16* Xf     = (__bf16*)carve((size_t)L_ * P_ * I_ * 2);  // 16MB
    __bf16* Hf     = (__bf16*)carve((size_t)P_ * H_ * 2);       // 512KB
    __bf16* FHf    = (__bf16*)carve((size_t)P_ * H_ * 2);
    __bf16* Vf     = (__bf16*)carve((size_t)P_ * H_ * 2);
    __bf16* h_row  = (__bf16*)carve((size_t)P_ * H_ * 2);       // bf16 h state
    float* c_state = (float*)carve((size_t)P_ * H_ * 4);
    float* final_h = (float*)carve((size_t)P_ * H_ * 4);
    float* v_out   = (float*)carve((size_t)P_ * H_ * 4);
    float* a_out   = (float*)carve((size_t)P_ * H_ * 4);
    float* pooled  = (float*)carve((size_t)H_ * 4);

    // state init (h0 = c0 = 0); async memset is graph-capture safe
    (void)hipMemsetAsync(Hf,      0, (size_t)P_ * H_ * 2, stream);
    (void)hipMemsetAsync(c_state, 0, (size_t)P_ * H_ * 4, stream);
    (void)hipMemsetAsync(final_h, 0, (size_t)P_ * H_ * 4, stream);

    // weight fragment packing (fp32 -> bf16 B-fragments)
    pack_b_kernel<<<512,  256, 0, stream>>>(W_ih, WihF, 16);                       // N=4096,K=512
    pack_b_kernel<<<1024, 256, 0, stream>>>(W_hh, WhhF, 32);                       // N=4096,K=1024
    pack_b_kernel<<<256,  256, 0, stream>>>(W_in + (size_t)2 * H_ * H_, WvF, 32);  // v rows of in_proj
    pack_b_kernel<<<256,  256, 0, stream>>>(W_out, WoutF, 32);                     // N=1024,K=1024

    // embedding gather -> bf16 A-fragments for all 64 steps
    embed_pack_kernel<<<2048, 256, 0, stream>>>(paths, emb, Xf);

    // 64 serialized LSTM steps: fused WMMA GEMM + cell; then repack h -> A-frags
    for (int t = 0; t < L_; ++t) {
        lstm_step_kernel<<<128, 256, 0, stream>>>(Xf, Hf, WihF, WhhF, b_ih, b_hh,
                                                  lengths, c_state, h_row, final_h, t);
        pack_a_bf16_kernel<<<64, 256, 0, stream>>>(h_row, Hf, 32);
    }

    // attention with seq_len==1: softmax==1 -> ctx == v projection
    pack_a_kernel<<<64, 256, 0, stream>>>(final_h, FHf, 32);
    gemm_bias_kernel<<<32, 256, 0, stream>>>(FHf, WvF, b_in + 2 * H_, v_out, H_);
    pack_a_kernel<<<64, 256, 0, stream>>>(v_out, Vf, 32);
    gemm_bias_kernel<<<32, 256, 0, stream>>>(Vf, WoutF, b_out, a_out, H_);

    // max over paths, then tiny sigmoid head
    maxpool_kernel<<<4, 256, 0, stream>>>(a_out, pooled);
    final_out_kernel<<<1, 128, 0, stream>>>(pooled, W_lin, b_lin, out);
}
